// LocalAttentionEncoderLayer_80573586473164
// MI455X (gfx1250) — matrix operand order repaired and checked by reading, not compiled, not run on hardware
//
#include <hip/hip_runtime.h>

// MI455X / gfx1250 fused local-attention encoder layer.
// Compute-bound (≈214 GFLOP vs ≈140MB HBM traffic) -> bf16 WMMA path,
// f32 accumulation, one 128-token window per workgroup, 320KB LDS.

#define DMODEL 512
#define NH 8
#define HDIM 64

#define SA 520      // padded stride (elems) for x / x1 / attn / ffacc
#define SQK 72      // padded stride for q,k (128x64)
#define SVT 136     // padded stride for v^T (64x128)
#define SP 136      // probs stride (128x128)
#define SH 136      // hidden chunk stride (128x128)

#define OFF_A 0
#define OFF_B 133120
#define OFF_C 266240
#define OFF_Q OFF_C
#define OFF_K (OFF_C + 18432)
#define OFF_VT (OFF_C + 36864)
#define OFF_P OFF_C
#define OFF_H OFF_C
#define SMEM_BYTES 320512

typedef __attribute__((ext_vector_type(16))) __bf16 bf16x16;
typedef __attribute__((ext_vector_type(8)))  float  f32x8;

static __device__ __forceinline__ unsigned short f2bf(float f) {
  union { float f; unsigned u; } v; v.f = f;
  unsigned r = v.u + 0x7FFFu + ((v.u >> 16) & 1u);   // round-to-nearest-even
  return (unsigned short)(r >> 16);
}
static __device__ __forceinline__ float bf2f(unsigned short h) {
  union { unsigned u; float f; } v; v.u = ((unsigned)h) << 16;
  return v.f;
}

union Frag32B { struct { uint4 lo, hi; } u; bf16x16 v; };

static __device__ __forceinline__ f32x8 splat8(float x) {
  f32x8 v;
#pragma unroll
  for (int i = 0; i < 8; ++i) v[i] = x;
  return v;
}

// A-fragment (16x32 bf16): lane holds row (row0+ln16); K = k0+half*16+{0..15}
// in elems 0..15 (two 16B LDS loads).
static __device__ __forceinline__ bf16x16 ld_a(const char* base, int strideE,
                                               int rowLane, int k0, int half) {
  const char* p = base + ((rowLane * strideE + k0 + half * 16) << 1);
  Frag32B f;
  f.u.lo = *(const uint4*)p;
  f.u.hi = *(const uint4*)(p + 16);
  return f.v;
}
// B-fragment (32x16 bf16) from row-major [N][K] LDS matrix: lane = col N,
// elems 0..15 = K = k0 + half*16 + {0..15}  (one contiguous 32B strip).
static __device__ __forceinline__ bf16x16 ld_b_lds(const char* base, int strideE,
                                                   int nrow, int k0, int half) {
  const char* p = base + ((nrow * strideE + k0 + half * 16) << 1);
  Frag32B f;
  f.u.lo = *(const uint4*)p;
  f.u.hi = *(const uint4*)(p + 16);
  return f.v;
}
// B-fragment from global row-major bf16 weights (row pointer pre-offset).
static __device__ __forceinline__ bf16x16 ld_b_g(const unsigned short* rowbase,
                                                 int k0, int half) {
  const uint4* p = (const uint4*)(rowbase + k0 + half * 16);
  Frag32B f; f.u.lo = p[0]; f.u.hi = p[1];
  return f.v;
}
static __device__ __forceinline__ f32x8 wmma_bf16(bf16x16 a, bf16x16 b, f32x8 c) {
  return __builtin_amdgcn_wmma_f32_16x16x32_bf16(false, a, false, b, (short)0, c,
                                                 false, false);
}

__global__ void cvt_bf16_kernel(const float* __restrict__ s,
                                unsigned short* __restrict__ d, int n) {
  int i = blockIdx.x * blockDim.x + threadIdx.x;
  if (i < n) d[i] = f2bf(s[i]);
}

__global__ __launch_bounds__(256, 1)
void encoder_layer_kernel(const float* __restrict__ src,
                          const unsigned short* __restrict__ wqkv,
                          const unsigned short* __restrict__ wo,
                          const unsigned short* __restrict__ w1,
                          const unsigned short* __restrict__ w2,
                          const float* __restrict__ in_b,
                          const float* __restrict__ out_b,
                          const float* __restrict__ ln1g,
                          const float* __restrict__ ln1b,
                          const float* __restrict__ b1,
                          const float* __restrict__ b2,
                          const float* __restrict__ ln2g,
                          const float* __restrict__ ln2b,
                          float* __restrict__ out) {
  extern __shared__ char smem[];
  const int tid  = threadIdx.x;
  const int wv   = tid >> 5;
  const int lane = tid & 31;
  const int half = lane >> 4;
  const int ln16 = lane & 15;
  const int win  = blockIdx.x;            // 256 windows (batch-major)
  const float* xg = src + (size_t)win * (128 * DMODEL);
  float* og       = out + (size_t)win * (128 * DMODEL);

  // ---- Phase A: x (f32) -> LDS bf16 (region A) -------------------------
  {
    const float4* x4 = (const float4*)xg;
#pragma unroll 4
    for (int i = tid; i < 128 * DMODEL / 4; i += 256) {
      int row = i >> 7, c4 = (i & 127) << 2;
      float4 v = x4[i];
      uint2 p;
      p.x = (unsigned)f2bf(v.x) | ((unsigned)f2bf(v.y) << 16);
      p.y = (unsigned)f2bf(v.z) | ((unsigned)f2bf(v.w) << 16);
      *(uint2*)(smem + OFF_A + ((row * SA + c4) << 1)) = p;
    }
  }
  __syncthreads();

  // ---- Phase B: attention, head by head --------------------------------
  for (int h = 0; h < NH; ++h) {
    // B1: q,k,v^T for this head. 96 tiles, 12 per wave, A shared per pair.
    for (int j = 0; j < 12; j += 2) {
      int t0 = wv * 12 + j;
      int which = t0 >> 5;                 // 0=q 1=k 2=v
      int rb = (t0 & 31) >> 2;
      int cb = t0 & 3;
      int col0 = h * HDIM + cb * 16 + ln16, col1 = col0 + 16;
      const unsigned short* wr0 = wqkv + (size_t)(which * DMODEL + col0) * DMODEL;
      const unsigned short* wr1 = wqkv + (size_t)(which * DMODEL + col1) * DMODEL;
      f32x8 a0 = splat8(0.f), a1 = splat8(0.f);
#pragma unroll
      for (int k0 = 0; k0 < DMODEL; k0 += 32) {
        bf16x16 af = ld_a(smem + OFF_A, SA, rb * 16 + ln16, k0, half);
        a0 = wmma_bf16(af, ld_b_g(wr0, k0, half), a0);
        a1 = wmma_bf16(af, ld_b_g(wr1, k0, half), a1);
      }
      float bi0 = in_b[which * DMODEL + col0];
      float bi1 = in_b[which * DMODEL + col1];
      if (which < 2) {
        char* base = smem + OFF_C + which * 18432;
#pragma unroll
        for (int r = 0; r < 8; ++r) {
          int row = rb * 16 + half * 8 + r;
          *(unsigned short*)(base + ((row * SQK + cb * 16 + ln16) << 1)) = f2bf(a0[r] + bi0);
          *(unsigned short*)(base + ((row * SQK + (cb + 1) * 16 + ln16) << 1)) = f2bf(a1[r] + bi1);
        }
      } else {                             // store v transposed: vT[d][token]
        char* base = smem + OFF_VT;
#pragma unroll
        for (int r = 0; r < 8; ++r) {
          int row = rb * 16 + half * 8 + r;
          *(unsigned short*)(base + (((cb * 16 + ln16) * SVT + row) << 1)) = f2bf(a0[r] + bi0);
          *(unsigned short*)(base + ((((cb + 1) * 16 + ln16) * SVT + row) << 1)) = f2bf(a1[r] + bi1);
        }
      }
    }
    __syncthreads();

    // B2: scores (wave owns 16 query rows x 128 keys = 8 tiles, K=64).
    f32x8 sc[8];
#pragma unroll
    for (int nb = 0; nb < 8; ++nb) sc[nb] = splat8(0.f);
#pragma unroll
    for (int k0 = 0; k0 < HDIM; k0 += 32) {
      bf16x16 af = ld_a(smem + OFF_Q, SQK, wv * 16 + ln16, k0, half);
#pragma unroll
      for (int nb = 0; nb < 8; ++nb) {
        bf16x16 bf = ld_b_lds(smem + OFF_K, SQK, nb * 16 + ln16, k0, half);
        sc[nb] = wmma_bf16(af, bf, sc[nb]);
      }
    }
    __syncthreads();                       // q/k reads done before probs overwrite

    // softmax in registers; rows m = half*8+r, cols spread over 16 lanes x 8 tiles
    float rmax[8], rinv[8];
#pragma unroll
    for (int r = 0; r < 8; ++r) {
      float m = -3.0e38f;
#pragma unroll
      for (int nb = 0; nb < 8; ++nb) { sc[nb][r] *= 0.125f; m = fmaxf(m, sc[nb][r]); }
#pragma unroll
      for (int d = 1; d < 16; d <<= 1) m = fmaxf(m, __shfl_xor(m, d, 32));
      rmax[r] = m;
    }
#pragma unroll
    for (int r = 0; r < 8; ++r) {
      float s = 0.f;
#pragma unroll
      for (int nb = 0; nb < 8; ++nb) { float e = __expf(sc[nb][r] - rmax[r]); sc[nb][r] = e; s += e; }
#pragma unroll
      for (int d = 1; d < 16; d <<= 1) s += __shfl_xor(s, d, 32);
      rinv[r] = 1.0f / s;
    }
#pragma unroll
    for (int nb = 0; nb < 8; ++nb)
#pragma unroll
      for (int r = 0; r < 8; ++r) {
        int row = wv * 16 + half * 8 + r;
        *(unsigned short*)(smem + OFF_P + ((row * SP + nb * 16 + ln16) << 1)) =
            f2bf(sc[nb][r] * rinv[r]);
      }
    __syncthreads();

    // B3: attn = probs @ v  (wave owns rows wv*16..+15, 4 col tiles, K=128)
    f32x8 av[4];
#pragma unroll
    for (int cb = 0; cb < 4; ++cb) av[cb] = splat8(0.f);
#pragma unroll
    for (int k0 = 0; k0 < 128; k0 += 32) {
      bf16x16 af = ld_a(smem + OFF_P, SP, wv * 16 + ln16, k0, half);
#pragma unroll
      for (int cb = 0; cb < 4; ++cb) {
        bf16x16 bf = ld_b_lds(smem + OFF_VT, SVT, cb * 16 + ln16, k0, half);
        av[cb] = wmma_bf16(af, bf, av[cb]);
      }
    }
#pragma unroll
    for (int cb = 0; cb < 4; ++cb)
#pragma unroll
      for (int r = 0; r < 8; ++r) {
        int row = wv * 16 + half * 8 + r;
        int col = h * HDIM + cb * 16 + ln16;
        *(unsigned short*)(smem + OFF_B + ((row * SA + col) << 1)) = f2bf(av[cb][r]);
      }
    __syncthreads();
  }

  // ---- Phase C: out-proj + residual + LN1 (wave owns rows wv*16..+15) --
  float rs[8], rq[8];
#pragma unroll
  for (int r = 0; r < 8; ++r) { rs[r] = 0.f; rq[r] = 0.f; }
  for (int ct = 0; ct < 32; ct += 2) {
    int c0 = ct * 16 + ln16, c1 = c0 + 16;
    const unsigned short* wr0 = wo + (size_t)c0 * DMODEL;
    const unsigned short* wr1 = wo + (size_t)c1 * DMODEL;
    f32x8 a0 = splat8(0.f), a1 = splat8(0.f);
#pragma unroll
    for (int k0 = 0; k0 < DMODEL; k0 += 32) {
      bf16x16 af = ld_a(smem + OFF_B, SA, wv * 16 + ln16, k0, half);
      a0 = wmma_bf16(af, ld_b_g(wr0, k0, half), a0);
      a1 = wmma_bf16(af, ld_b_g(wr1, k0, half), a1);
    }
    float bo0 = out_b[c0], bo1 = out_b[c1];
#pragma unroll
    for (int r = 0; r < 8; ++r) {
      int row = wv * 16 + half * 8 + r;
      float v0 = a0[r] + bo0 + xg[row * DMODEL + c0];
      float v1 = a1[r] + bo1 + xg[row * DMODEL + c1];
      rs[r] += v0 + v1; rq[r] += v0 * v0 + v1 * v1;
      *(unsigned short*)(smem + OFF_A + ((row * SA + c0) << 1)) = f2bf(v0);
      *(unsigned short*)(smem + OFF_A + ((row * SA + c1) << 1)) = f2bf(v1);
    }
  }
  float mean[8], rstd[8];
#pragma unroll
  for (int r = 0; r < 8; ++r) {
#pragma unroll
    for (int d = 1; d < 16; d <<= 1) { rs[r] += __shfl_xor(rs[r], d, 32); rq[r] += __shfl_xor(rq[r], d, 32); }
    mean[r] = rs[r] * (1.0f / DMODEL);
    rstd[r] = rsqrtf(rq[r] * (1.0f / DMODEL) - mean[r] * mean[r] + 1e-5f);
  }
  for (int ct = 0; ct < 32; ++ct) {
    int c0 = ct * 16 + ln16;
    float g = ln1g[c0], bb = ln1b[c0];
#pragma unroll
    for (int r = 0; r < 8; ++r) {
      int row = wv * 16 + half * 8 + r;
      char* p = smem + OFF_A + ((row * SA + c0) << 1);
      float v = bf2f(*(unsigned short*)p);
      *(unsigned short*)p = f2bf((v - mean[r]) * rstd[r] * g + bb);
    }
  }
  // No barrier: FFN is row-partitioned identically per wave.

  // ---- Phase D: FFN, 16 hidden chunks of 128 (Hc in region C) ----------
  for (int c = 0; c < 16; ++c) {
    // D1: Hc = relu(x1 @ W1_chunk^T), wave row block, 8 col tiles, K=512
    for (int cb = 0; cb < 8; cb += 2) {
      int f0 = c * 128 + cb * 16 + ln16, f1 = f0 + 16;
      const unsigned short* wr0 = w1 + (size_t)f0 * DMODEL;
      const unsigned short* wr1 = w1 + (size_t)f1 * DMODEL;
      f32x8 a0 = splat8(0.f), a1 = splat8(0.f);
#pragma unroll
      for (int k0 = 0; k0 < DMODEL; k0 += 32) {
        bf16x16 af = ld_a(smem + OFF_A, SA, wv * 16 + ln16, k0, half);
        a0 = wmma_bf16(af, ld_b_g(wr0, k0, half), a0);
        a1 = wmma_bf16(af, ld_b_g(wr1, k0, half), a1);
      }
      float bi0 = b1[f0], bi1 = b1[f1];
#pragma unroll
      for (int r = 0; r < 8; ++r) {
        int row = wv * 16 + half * 8 + r;
        *(unsigned short*)(smem + OFF_H + ((row * SH + cb * 16 + ln16) << 1)) = f2bf(fmaxf(a0[r] + bi0, 0.f));
        *(unsigned short*)(smem + OFF_H + ((row * SH + (cb + 1) * 16 + ln16) << 1)) = f2bf(fmaxf(a1[r] + bi1, 0.f));
      }
    }
    // D2: ffacc += Hc @ W2_chunk^T (accumulate through WMMA C operand)
    for (int ct = 0; ct < 32; ct += 2) {
      int d0 = ct * 16 + ln16, d1 = d0 + 16;
      const unsigned short* wr0 = w2 + (size_t)d0 * 2048 + c * 128;
      const unsigned short* wr1 = w2 + (size_t)d1 * 2048 + c * 128;
      f32x8 c0v, c1v;
      if (c == 0) { c0v = splat8(b2[d0]); c1v = splat8(b2[d1]); }
      else {
#pragma unroll
        for (int r = 0; r < 8; ++r) {
          int row = wv * 16 + half * 8 + r;
          c0v[r] = bf2f(*(unsigned short*)(smem + OFF_B + ((row * SA + d0) << 1)));
          c1v[r] = bf2f(*(unsigned short*)(smem + OFF_B + ((row * SA + d1) << 1)));
        }
      }
#pragma unroll
      for (int k0 = 0; k0 < 128; k0 += 32) {
        bf16x16 af = ld_a(smem + OFF_H, SH, wv * 16 + ln16, k0, half);
        c0v = wmma_bf16(af, ld_b_g(wr0, k0, half), c0v);
        c1v = wmma_bf16(af, ld_b_g(wr1, k0, half), c1v);
      }
#pragma unroll
      for (int r = 0; r < 8; ++r) {
        int row = wv * 16 + half * 8 + r;
        *(unsigned short*)(smem + OFF_B + ((row * SA + d0) << 1)) = f2bf(c0v[r]);
        *(unsigned short*)(smem + OFF_B + ((row * SA + d1) << 1)) = f2bf(c1v[r]);
      }
    }
  }
  __syncthreads();

  // ---- Phase E: out = LN2(x1 + ff) -------------------------------------
  {
    int row = tid >> 1;
    int cb = (tid & 1) * 256;
    const char* xr = smem + OFF_A + row * (SA * 2);
    const char* fr = smem + OFF_B + row * (SA * 2);
    float s = 0.f, s2 = 0.f;
    for (int col = cb; col < cb + 256; col += 2) {
      unsigned a = *(const unsigned*)(xr + (col << 1));
      unsigned b = *(const unsigned*)(fr + (col << 1));
      float v0 = bf2f((unsigned short)a) + bf2f((unsigned short)b);
      float v1 = bf2f((unsigned short)(a >> 16)) + bf2f((unsigned short)(b >> 16));
      s += v0 + v1; s2 += v0 * v0 + v1 * v1;
    }
    s += __shfl_xor(s, 1, 32);
    s2 += __shfl_xor(s2, 1, 32);
    float mu = s * (1.0f / DMODEL);
    float rst = rsqrtf(s2 * (1.0f / DMODEL) - mu * mu + 1e-5f);
    for (int col = cb; col < cb + 256; col += 2) {
      unsigned a = *(const unsigned*)(xr + (col << 1));
      unsigned b = *(const unsigned*)(fr + (col << 1));
      float v0 = bf2f((unsigned short)a) + bf2f((unsigned short)b);
      float v1 = bf2f((unsigned short)(a >> 16)) + bf2f((unsigned short)(b >> 16));
      float2 o;
      o.x = (v0 - mu) * rst * ln2g[col] + ln2b[col];
      o.y = (v1 - mu) * rst * ln2g[col + 1] + ln2b[col + 1];
      *(float2*)(og + row * DMODEL + col) = o;
    }
  }
}

extern "C" void kernel_launch(void* const* d_in, const int* in_sizes, int n_in,
                              void* d_out, int out_size, void* d_ws, size_t ws_size,
                              hipStream_t stream) {
  (void)in_sizes; (void)n_in; (void)out_size; (void)ws_size;
  const float* src    = (const float*)d_in[0];
  const float* wqkv_f = (const float*)d_in[1];
  const float* in_b   = (const float*)d_in[2];
  const float* wo_f   = (const float*)d_in[3];
  const float* out_b  = (const float*)d_in[4];
  const float* ln1g   = (const float*)d_in[5];
  const float* ln1b   = (const float*)d_in[6];
  const float* w1_f   = (const float*)d_in[7];
  const float* b1     = (const float*)d_in[8];
  const float* w2_f   = (const float*)d_in[9];
  const float* b2     = (const float*)d_in[10];
  const float* ln2g   = (const float*)d_in[11];
  const float* ln2b   = (const float*)d_in[12];
  float* out = (float*)d_out;

  unsigned short* ws   = (unsigned short*)d_ws;
  unsigned short* wqkv = ws;                 // 1536*512
  unsigned short* wo   = ws + 786432;        // 512*512
  unsigned short* w1   = ws + 1048576;       // 2048*512
  unsigned short* w2   = ws + 2097152;       // 512*2048

  cvt_bf16_kernel<<<(786432 + 255) / 256, 256, 0, stream>>>(wqkv_f, wqkv, 786432);
  cvt_bf16_kernel<<<(262144 + 255) / 256, 256, 0, stream>>>(wo_f, wo, 262144);
  cvt_bf16_kernel<<<(1048576 + 255) / 256, 256, 0, stream>>>(w1_f, w1, 1048576);
  cvt_bf16_kernel<<<(1048576 + 255) / 256, 256, 0, stream>>>(w2_f, w2, 1048576);

  hipFuncSetAttribute((const void*)encoder_layer_kernel,
                      hipFuncAttributeMaxDynamicSharedMemorySize, SMEM_BYTES);
  encoder_layer_kernel<<<256, 256, SMEM_BYTES, stream>>>(
      src, wqkv, wo, w1, w2, in_b, out_b, ln1g, ln1b, b1, b2, ln2g, ln2b, out);
}
